// VSSBlock_51393578664452
// MI455X (gfx1250) — compile-verified, matrix operand order checked by reading, not run on hardware
//
#include <hip/hip_runtime.h>
#include <hip/hip_bf16.h>

typedef __attribute__((ext_vector_type(16))) _Float16 v16h;
typedef __attribute__((ext_vector_type(8)))  _Float16 v8h;
typedef __attribute__((ext_vector_type(8)))  float    v8f;

#define DEV __device__ __forceinline__

// Flip to 0 if the gfx1250 assembler rejects the async-to-LDS mnemonics.
#define USE_ASYNC_COPY 1

constexpr int Bb = 4, DM = 128, DI = 256, L = 4096;
constexpr int BLT = Bb * L;            // 16384 tokens
constexpr int DSTATE = 4, DTRANK = 16;
constexpr int NB = 64;                 // N-block per workgroup (4 tiles/wave)
constexpr int LDS_PITCH_PAD = 8;       // +8 halves => 528B pitch, bank stride 4

DEV float sigmoidf_(float x){ return 1.f / (1.f + __expf(-x)); }
DEV float siluf_(float x){ return x * sigmoidf_(x); }

// ---------------- WMMA fragment loader (wave32, 16x16x32 f16) ---------------
// lane&15 = row; K-half by lane>>4: elems 0..7 -> K=base..base+7,
// elems 8..15 -> K=base+16..base+23, base=(lane>>4)*8. Two 16B loads.
// Works for both global (global_load_b128) and LDS (ds_load_b128) pointers.
DEV v16h load_frag(const _Float16* base, int ld, int r0, int k0){
  int lane = threadIdx.x & 31;
  const _Float16* p = base + (size_t)(r0 + (lane & 15)) * ld + (k0 + ((lane >> 4) << 3));
  v8h lo = *reinterpret_cast<const v8h*>(p);
  v8h hi = *reinterpret_cast<const v8h*>(p + 16);
  v16h v;
#pragma unroll
  for (int i = 0; i < 8; ++i){ v[i] = lo[i]; v[i + 8] = hi[i]; }
  return v;
}

// Cooperative stage of a 64-row x K weight slab into LDS (padded pitch).
// Uses CDNA5 async global->LDS copies tracked by ASYNCcnt.
DEV void stage_weights(_Float16* sW, const _Float16* Wb, int K){
  int tid = threadIdx.x;
  int cpr = K >> 3;                 // 16-byte chunks per row
  int total = NB * cpr;             // divisible by 256 for K in {128,256}
  int P = K + LDS_PITCH_PAD;
#if USE_ASYNC_COPY
  uint32_t sbase = (uint32_t)(uintptr_t)sW;   // low 32 bits = LDS byte offset
  for (int c = tid; c < total; c += 256){
    int row = c / cpr, col = c - row * cpr;
    uint32_t loff = sbase + (uint32_t)(row * P + col * 8) * 2u;
    uint64_t g = (uint64_t)(uintptr_t)(Wb + (size_t)row * K + col * 8);
    asm volatile("global_load_async_to_lds_b128 %0, %1, off"
                 :: "v"(loff), "v"(g) : "memory");
  }
  asm volatile("s_wait_asynccnt 0x0" ::: "memory");
#else
  for (int c = tid; c < total; c += 256){
    int row = c / cpr, col = c - row * cpr;
    *reinterpret_cast<v8h*>(sW + row * P + col * 8) =
        *reinterpret_cast<const v8h*>(Wb + (size_t)row * K + col * 8);
  }
#endif
  __syncthreads();
}

// One wave computes a 16x64 strip: 4 accumulators share each A fragment.
DEV void wmma_strip(const _Float16* A, const _Float16* sW, int K, int row0,
                    v8f acc[4]){
  int lane = threadIdx.x & 31;
  int P = K + LDS_PITCH_PAD;
  for (int k = 0; k < K; k += 32){
    if (k + 32 < K)
      __builtin_prefetch(A + (size_t)(row0 + (lane & 15)) * K + k + 32, 0, 1);
    v16h a = load_frag(A, K, row0, k);
#pragma unroll
    for (int j = 0; j < 4; ++j){
      v16h b = load_frag(sW, P, j * 16, k);
      acc[j] = __builtin_amdgcn_wmma_f32_16x16x32_f16(false, a, false, b,
                                                      (short)0, acc[j],
                                                      false, false);
    }
  }
}

DEV void strip_pos(int& row0, int& n0){
  int wave = threadIdx.x >> 5;
  row0 = blockIdx.y * 128 + wave * 16;
  n0   = blockIdx.x * NB;
}

#define GEMM_PROLOGUE(Kv)                                                    \
  __shared__ _Float16 sWt[NB * (256 + LDS_PITCH_PAD)];                       \
  int row0, n0; strip_pos(row0, n0);                                         \
  stage_weights(sWt, W + (size_t)n0 * (Kv), (Kv));                           \
  v8f acc[4] = {{0,0,0,0,0,0,0,0},{0,0,0,0,0,0,0,0},                         \
                {0,0,0,0,0,0,0,0},{0,0,0,0,0,0,0,0}};                        \
  wmma_strip(A, sWt, (Kv), row0, acc);                                       \
  int lane = threadIdx.x & 31;

// ---------------- GEMM kernels (grid: (N/64, M/128, Z), block 256) ----------
__global__ void k_gemm_f16(const _Float16* A, size_t sA, const _Float16* W, size_t sW,
                           _Float16* C, size_t sC, int K, int ldc){
  int z = blockIdx.z;
  A += (size_t)z * sA; W += (size_t)z * sW; C += (size_t)z * sC;
  GEMM_PROLOGUE(K)
#pragma unroll
  for (int j = 0; j < 4; ++j){
    int n = n0 + j * 16 + (lane & 15);
#pragma unroll
    for (int r = 0; r < 8; ++r){
      int m = row0 + r + ((lane >> 4) << 3);
      C[(size_t)m * ldc + n] = (_Float16)acc[j][r];
    }
  }
}

__global__ void k_gemm_f32(const _Float16* A, size_t sA, const _Float16* W, size_t sW,
                           float* C, size_t sC, int K, int ldc){
  int z = blockIdx.z;
  A += (size_t)z * sA; W += (size_t)z * sW; C += (size_t)z * sC;
  GEMM_PROLOGUE(K)
#pragma unroll
  for (int j = 0; j < 4; ++j){
    int n = n0 + j * 16 + (lane & 15);
#pragma unroll
    for (int r = 0; r < 8; ++r){
      int m = row0 + r + ((lane >> 4) << 3);
      C[(size_t)m * ldc + n] = acc[j][r];
    }
  }
}

// out_proj: x2 = hn @ Wop^T + x_flat (x_flat gathered from NCHW input)
__global__ void k_gemm_outproj(const _Float16* A, const _Float16* W, const float* x,
                               float* x2, int K){
  GEMM_PROLOGUE(K)
#pragma unroll
  for (int j = 0; j < 4; ++j){
    int n = n0 + j * 16 + (lane & 15);
#pragma unroll
    for (int r = 0; r < 8; ++r){
      int m = row0 + r + ((lane >> 4) << 3);
      int b = m >> 12, l = m & 4095;
      x2[(size_t)m * DM + n] = acc[j][r] + x[(((size_t)b * DM + n) << 12) + l];
    }
  }
}

// ffn1: f1 = gelu(x2n @ W1^T) (exact erf GELU)
__global__ void k_gemm_gelu(const _Float16* A, const _Float16* W, _Float16* C, int K){
  GEMM_PROLOGUE(K)
#pragma unroll
  for (int j = 0; j < 4; ++j){
    int n = n0 + j * 16 + (lane & 15);
#pragma unroll
    for (int r = 0; r < 8; ++r){
      int m = row0 + r + ((lane >> 4) << 3);
      float v = acc[j][r];
      C[(size_t)m * DM + n] =
          (_Float16)(0.5f * v * (1.f + erff(v * 0.70710678118654752f)));
    }
  }
}

// ffn2: out(NCHW) = f1 @ W2^T + x2
__global__ void k_gemm_final(const _Float16* A, const _Float16* W, const float* x2,
                             float* out, int K){
  GEMM_PROLOGUE(K)
#pragma unroll
  for (int j = 0; j < 4; ++j){
    int n = n0 + j * 16 + (lane & 15);
#pragma unroll
    for (int r = 0; r < 8; ++r){
      int m = row0 + r + ((lane >> 4) << 3);
      int b = m >> 12, l = m & 4095;
      out[(((size_t)b * DM + n) << 12) + l] = acc[j][r] + x2[(size_t)m * DM + n];
    }
  }
}

// ---------------- elementwise / norm / scan kernels -------------------------
__global__ void k_f32_to_f16(const float* s, _Float16* d, int n){
  int i = blockIdx.x * 256 + threadIdx.x;
  if (i < n) d[i] = (_Float16)s[i];
}

__global__ void k_ln1(const float* x, const float* g, const float* bt, _Float16* o){
  int m = blockIdx.x, c = threadIdx.x;
  int b = m >> 12, l = m & 4095;
  float v = x[(((size_t)b * DM + c) << 12) + l];
  __shared__ float red[DM];
  red[c] = v; __syncthreads();
  for (int s = DM / 2; s > 0; s >>= 1){ if (c < s) red[c] += red[c + s]; __syncthreads(); }
  float mean = red[0] * (1.f / DM); __syncthreads();
  float d = v - mean; red[c] = d * d; __syncthreads();
  for (int s = DM / 2; s > 0; s >>= 1){ if (c < s) red[c] += red[c + s]; __syncthreads(); }
  float var = red[0] * (1.f / DM);
  o[(size_t)m * DM + c] = (_Float16)(d * rsqrtf(var + 1e-5f) * g[c] + bt[c]);
}

__global__ void k_ln2(const float* x, const float* g, const float* bt, _Float16* o){
  int m = blockIdx.x, c = threadIdx.x;
  float v = x[(size_t)m * DM + c];
  __shared__ float red[DM];
  red[c] = v; __syncthreads();
  for (int s = DM / 2; s > 0; s >>= 1){ if (c < s) red[c] += red[c + s]; __syncthreads(); }
  float mean = red[0] * (1.f / DM); __syncthreads();
  float d = v - mean; red[c] = d * d; __syncthreads();
  for (int s = DM / 2; s > 0; s >>= 1){ if (c < s) red[c] += red[c + s]; __syncthreads(); }
  float var = red[0] * (1.f / DM);
  o[(size_t)m * DM + c] = (_Float16)(d * rsqrtf(var + 1e-5f) * g[c] + bt[c]);
}

// 3x3 depthwise conv + SiLU, scatter into the 4 scan orders (involutions).
__global__ void k_dwconv(const _Float16* hin, const float* w, _Float16* hs){
  int m = blockIdx.x, d = threadIdx.x;
  int b = m >> 12, l = m & 4095, hh = l >> 6, ww = l & 63;
  float acc = 0.f;
#pragma unroll
  for (int i = 0; i < 3; ++i){
    int yy = hh + i - 1; if ((unsigned)yy >= 64u) continue;
#pragma unroll
    for (int j = 0; j < 3; ++j){
      int xx = ww + j - 1; if ((unsigned)xx >= 64u) continue;
      acc += w[d * 9 + i * 3 + j] *
             (float)hin[(size_t)(((b << 12) | (yy << 6) | xx)) * DI + d];
    }
  }
  _Float16 hv = (_Float16)siluf_(acc);
  int T = ((l & 63) << 6) | (l >> 6);
  size_t bL = (size_t)b * L;
  hs[((size_t)0 * BLT + bL + l)           * DI + d] = hv;
  hs[((size_t)1 * BLT + bL + (L - 1 - l)) * DI + d] = hv;
  hs[((size_t)2 * BLT + bL + T)           * DI + d] = hv;
  hs[((size_t)3 * BLT + bL + (L - 1 - T)) * DI + d] = hv;
}

// causal depthwise conv1d (K=4) + bias + SiLU on the xm half of xz
__global__ void k_conv1d(const _Float16* xz, const float* w, const float* bias,
                         _Float16* xm){
  int bl = blockIdx.x, dir = blockIdx.y, d = threadIdx.x;
  int s = bl & 4095;
  float acc = bias[dir * DI + d];
#pragma unroll
  for (int j = 0; j < 4; ++j){
    int sl = s - 3 + j; if (sl < 0) continue;
    acc += w[(dir * DI + d) * 4 + j] *
           (float)xz[((size_t)dir * BLT + (bl - 3 + j)) * (2 * DI) + d];
  }
  xm[((size_t)dir * BLT + bl) * DI + d] = (_Float16)siluf_(acc);
}

__global__ void k_xproj(const _Float16* xm, const float* xw, float* xdbl){
  int bl = blockIdx.x, dir = blockIdx.y, t = threadIdx.x;
  __shared__ float sx[DI];
  sx[t] = (float)xm[((size_t)dir * BLT + bl) * DI + t];
  __syncthreads();
  if (t < 24){
    const float* wr = xw + ((size_t)dir * 24 + t) * DI;
    float acc = 0.f;
    for (int d = 0; d < DI; ++d) acc += sx[d] * wr[d];
    xdbl[((size_t)dir * BLT + bl) * 24 + t] = acc;
  }
}

__global__ void k_dt(const float* xdbl, const float* dtw, const float* dtb, float* dt){
  int bl = blockIdx.x, dir = blockIdx.y, d = threadIdx.x;
  __shared__ float xd[DTRANK];
  if (d < DTRANK) xd[d] = xdbl[((size_t)dir * BLT + bl) * 24 + d];
  __syncthreads();
  float acc = dtb[dir * DI + d];
#pragma unroll
  for (int r = 0; r < DTRANK; ++r) acc += xd[r] * dtw[((size_t)dir * DI + d) * DTRANK + r];
  dt[((size_t)dir * BLT + bl) * DI + d] = (acc > 20.f) ? acc : log1pf(__expf(acc));
}

// selective scan: 4 dirs x 4 batches x 256 channels concurrent (128 waves)
__global__ void k_scan(const float* dt, const float* xdbl, const _Float16* xm,
                       const _Float16* xz, const float* A_log, const float* Dp,
                       _Float16* yg){
  int b = blockIdx.x, dir = blockIdx.y, d = threadIdx.x;
  float A[DSTATE];
#pragma unroll
  for (int n = 0; n < DSTATE; ++n)
    A[n] = -__expf(A_log[((size_t)dir * DI + d) * DSTATE + n]);
  float Dv = Dp[dir * DI + d];
  float h[DSTATE] = {0.f, 0.f, 0.f, 0.f};
  size_t base = (size_t)dir * BLT + (size_t)b * L;
  for (int l = 0; l < L; ++l){
    size_t row = base + l;
    float dtv = dt[row * DI + d];
    float xv  = (float)xm[row * DI + d];
    const float* xd = xdbl + row * 24;
    float y = 0.f;
#pragma unroll
    for (int n = 0; n < DSTATE; ++n){
      float dA = __expf(dtv * A[n]);
      h[n] = dA * h[n] + dtv * xd[16 + n] * xv;
      y += h[n] * xd[20 + n];
    }
    y += Dv * xv;
    float zv = (float)xz[row * (2 * DI) + DI + d];
    yg[row * DI + d] = (_Float16)(y * siluf_(zv));
  }
}

// gather 4 directions (involution maps), /4, LN over 256 -> f16
__global__ void k_combine_ln(const float* yd, const float* g, const float* bt,
                             _Float16* hn){
  int m = blockIdx.x, d = threadIdx.x;
  int b = m >> 12, l = m & 4095;
  int T = ((l & 63) << 6) | (l >> 6);
  size_t bL = (size_t)b * L;
  float v = yd[((size_t)0 * BLT + bL + l)           * DI + d]
          + yd[((size_t)1 * BLT + bL + (L - 1 - l)) * DI + d]
          + yd[((size_t)2 * BLT + bL + T)           * DI + d]
          + yd[((size_t)3 * BLT + bL + (L - 1 - T)) * DI + d];
  v *= 0.25f;
  __shared__ float red[DI];
  red[d] = v; __syncthreads();
  for (int s = DI / 2; s > 0; s >>= 1){ if (d < s) red[d] += red[d + s]; __syncthreads(); }
  float mean = red[0] * (1.f / DI); __syncthreads();
  float dv = v - mean; red[d] = dv * dv; __syncthreads();
  for (int s = DI / 2; s > 0; s >>= 1){ if (d < s) red[d] += red[d + s]; __syncthreads(); }
  float var = red[0] * (1.f / DI);
  hn[(size_t)m * DI + d] = (_Float16)(dv * rsqrtf(var + 1e-5f) * g[d] + bt[d]);
}

// ---------------- host launcher --------------------------------------------
extern "C" void kernel_launch(void* const* d_in, const int* in_sizes, int n_in,
                              void* d_out, int out_size, void* d_ws, size_t ws_size,
                              hipStream_t stream){
  (void)in_sizes; (void)n_in; (void)out_size; (void)ws_size;
  const float* x        = (const float*)d_in[0];
  const float* n1g      = (const float*)d_in[1];
  const float* n1b      = (const float*)d_in[2];
  const float* in_proj  = (const float*)d_in[3];
  const float* dwconv_w = (const float*)d_in[4];
  const float* m_in_w   = (const float*)d_in[5];
  const float* m_conv_w = (const float*)d_in[6];
  const float* m_conv_b = (const float*)d_in[7];
  const float* m_xproj  = (const float*)d_in[8];
  const float* m_dt_w   = (const float*)d_in[9];
  const float* m_dt_b   = (const float*)d_in[10];
  const float* m_A_log  = (const float*)d_in[11];
  const float* m_D      = (const float*)d_in[12];
  const float* m_out_w  = (const float*)d_in[13];
  const float* nssg     = (const float*)d_in[14];
  const float* nssb     = (const float*)d_in[15];
  const float* out_proj = (const float*)d_in[16];
  const float* n2g      = (const float*)d_in[17];
  const float* n2b      = (const float*)d_in[18];
  const float* ffn_w1   = (const float*)d_in[19];
  const float* ffn_w2   = (const float*)d_in[20];
  float* out = (float*)d_out;

  char* p = (char*)d_ws;
  auto alloc = [&](size_t bytes)->char*{
    char* r = p; p += (bytes + 255) & ~(size_t)255; return r;
  };
  _Float16* wInP = (_Float16*)alloc((size_t)DI * DM * 2);
  _Float16* wMin = (_Float16*)alloc((size_t)4 * 2 * DI * DI * 2);
  _Float16* wMo  = (_Float16*)alloc((size_t)4 * DI * DI * 2);
  _Float16* wOp  = (_Float16*)alloc((size_t)DM * DI * 2);
  _Float16* wF1  = (_Float16*)alloc((size_t)DM * DM * 2);
  _Float16* wF2  = (_Float16*)alloc((size_t)DM * DM * 2);
  _Float16* x1h  = (_Float16*)alloc((size_t)BLT * DM * 2);
  _Float16* hbuf = (_Float16*)alloc((size_t)BLT * DI * 2);
  _Float16* hs   = (_Float16*)alloc((size_t)4 * BLT * DI * 2);
  _Float16* xzb  = (_Float16*)alloc((size_t)4 * BLT * 2 * DI * 2);
  _Float16* xmc  = (_Float16*)alloc((size_t)4 * BLT * DI * 2);
  float*    xdbl = (float*)   alloc((size_t)4 * BLT * 24 * 4);
  float*    dtb  = (float*)   alloc((size_t)4 * BLT * DI * 4);
  _Float16* ygb  = (_Float16*)alloc((size_t)4 * BLT * DI * 2);
  float*    ydir = (float*)   alloc((size_t)4 * BLT * DI * 4);
  _Float16* hnb  = (_Float16*)alloc((size_t)BLT * DI * 2);
  float*    x2b  = (float*)   alloc((size_t)BLT * DM * 4);
  _Float16* x2n  = (_Float16*)alloc((size_t)BLT * DM * 2);
  _Float16* f1b  = (_Float16*)alloc((size_t)BLT * DM * 2);

  auto cvt = [&](const float* s, _Float16* d, int n){
    k_f32_to_f16<<<(n + 255) / 256, 256, 0, stream>>>(s, d, n);
  };
  cvt(in_proj,  wInP, DI * DM);
  cvt(m_in_w,   wMin, 4 * 2 * DI * DI);
  cvt(m_out_w,  wMo,  4 * DI * DI);
  cvt(out_proj, wOp,  DM * DI);
  cvt(ffn_w1,   wF1,  DM * DM);
  cvt(ffn_w2,   wF2,  DM * DM);

  // LN1 -> in_proj GEMM -> dwconv+SiLU+scatter into 4 scan orders
  k_ln1<<<BLT, DM, 0, stream>>>(x, n1g, n1b, x1h);
  k_gemm_f16<<<dim3(DI / NB, BLT / 128, 1), 256, 0, stream>>>(
      x1h, 0, wInP, 0, hbuf, 0, DM, DI);
  k_dwconv<<<BLT, DI, 0, stream>>>(hbuf, dwconv_w, hs);

  // per-direction inner projection (all 4 dirs via grid z)
  k_gemm_f16<<<dim3(2 * DI / NB, BLT / 128, 4), 256, 0, stream>>>(
      hs, (size_t)BLT * DI, wMin, (size_t)2 * DI * DI,
      xzb, (size_t)BLT * 2 * DI, DI, 2 * DI);
  k_conv1d<<<dim3(BLT, 4), DI, 0, stream>>>(xzb, m_conv_w, m_conv_b, xmc);
  k_xproj<<<dim3(BLT, 4), DI, 0, stream>>>(xmc, m_xproj, xdbl);
  k_dt<<<dim3(BLT, 4), DI, 0, stream>>>(xdbl, m_dt_w, m_dt_b, dtb);

  // selective scan (all dirs/batches/channels concurrent)
  k_scan<<<dim3(Bb, 4), DI, 0, stream>>>(dtb, xdbl, xmc, xzb, m_A_log, m_D, ygb);

  // per-direction output GEMM, then gather+combine+LN
  k_gemm_f32<<<dim3(DI / NB, BLT / 128, 4), 256, 0, stream>>>(
      ygb, (size_t)BLT * DI, wMo, (size_t)DI * DI,
      ydir, (size_t)BLT * DI, DI, DI);
  k_combine_ln<<<BLT, DI, 0, stream>>>(ydir, nssg, nssb, hnb);

  // out_proj + residual, LN2, FFN (GELU), final residual to NCHW output
  k_gemm_outproj<<<dim3(DM / NB, BLT / 128, 1), 256, 0, stream>>>(hnb, wOp, x, x2b, DI);
  k_ln2<<<BLT, DM, 0, stream>>>(x2b, n2g, n2b, x2n);
  k_gemm_gelu<<<dim3(DM / NB, BLT / 128, 1), 256, 0, stream>>>(x2n, wF1, f1b, DM);
  k_gemm_final<<<dim3(DM / NB, BLT / 128, 1), 256, 0, stream>>>(f1b, wF2, x2b, out, DM);
}
